// POSSM_76536317215056
// MI455X (gfx1250) — matrix operand order repaired and checked by reading, compiled, not run on hardware
//
#include <hip/hip_runtime.h>

// ---------------------------------------------------------------------------
// POSSM forward on gfx1250 (CDNA5). Dense GEMMs: v_wmma_f32_16x16x32_f16 with
// LDS tiles staged via global_load_async_to_lds_b128 (ASYNCcnt). Weights are
// pre-transposed to [N,K] f16 so both A and B tiles stage as contiguous b128
// copies. Attention / LN / rope / GRU recurrence are VALU kernels. wave32.
// ---------------------------------------------------------------------------

typedef __attribute__((ext_vector_type(16))) _Float16 v16h;
typedef __attribute__((ext_vector_type(8)))  _Float16 v8h;
typedef __attribute__((ext_vector_type(8)))  float    v8f;
typedef __attribute__((address_space(3)))    _Float16 lds_h;

#define PB 8      // batch
#define PT 20     // time bins
#define PBT 160   // PB*PT
#define PNIN 256  // spikes per bin
#define PNL 8     // latents
#define PNOUT 128
#define PDIM 512
#define PRNN 512

// --------------------- async global->LDS staging helpers -------------------
__device__ __forceinline__ void async_copy_b128(_Float16* dst,
                                                const _Float16* src) {
  unsigned ldsoff = (unsigned)(size_t)(lds_h*)dst;  // LDS byte offset
  asm volatile("global_load_async_to_lds_b128 %0, %1, off" ::"v"(ldsoff),
               "v"(src)
               : "memory");
}
__device__ __forceinline__ void wait_async() {
  asm volatile("s_wait_asynccnt 0x0" ::: "memory");
}

// ------------------------------ rope helper --------------------------------
// rotate_dim = DH/2 = 32 pairs; only first 16 pair-frequencies are nonzero.
// inv[p] = 2*pi / (T_MIN * (T_MAX/T_MIN)^(p/16)),  T_MIN=1e-4, T_MAX/T_MIN=1e5
__device__ __forceinline__ void rope_cs(float t, int p, float& c, float& s) {
  if (p < 16) {
    float inv = 6.283185307179586f / (1e-4f * __powf(1e5f, (float)p * 0.0625f));
    float a = t * inv;
    s = sinf(a);
    c = cosf(a);
  } else {
    c = 1.0f;
    s = 0.0f;
  }
}

// ------------------------- generic WMMA GEMM kernel ------------------------
// C[M,N] = epilogue(A[M,K] @ W[K,N] + bias (+resid)); A f16 [M,K] row-major,
// WT f16 = W transposed, [N,K] row-major. f32 accumulate. Block = 256 threads
// = 8 waves, tile 128x128, BK=32. N%128==0, K%32==0; M guarded.
template <int OUT_F16, int GELU, int RESID>
__global__ __launch_bounds__(256) void gemm_wmma(
    const _Float16* __restrict__ A, const _Float16* __restrict__ WT,
    const float* __restrict__ bias, const float* __restrict__ resid,
    void* __restrict__ Cout, int M, int N, int K) {
  __shared__ _Float16 lsa[128 * 32];  // [row][k]
  __shared__ _Float16 lsb[128 * 32];  // [col][k]
  const int tid = threadIdx.x;
  const int lane = tid & 31, wave = tid >> 5;
  const int wm = wave >> 1;  // 0..3 : 32-row strip
  const int wn = wave & 1;   // 0..1 : 64-col strip
  const int bm = blockIdx.x * 128, bn = blockIdx.y * 128;

  v8f acc[2][4];
#pragma unroll
  for (int i = 0; i < 2; ++i)
#pragma unroll
    for (int j = 0; j < 4; ++j)
#pragma unroll
      for (int v = 0; v < 8; ++v) acc[i][j][v] = 0.0f;

  const int srow = tid >> 1;            // 0..127
  const int sseg = (tid & 1) * 16;      // 0 or 16 halves
  for (int k0 = 0; k0 < K; k0 += 32) {
    // stage A tile (async copy; zero-pad rows >= M)
    {
      _Float16* d0 = &lsa[srow * 32 + sseg];
      int grow = bm + srow;
      if (grow < M) {
        const _Float16* s0 = A + (size_t)grow * K + k0 + sseg;
        async_copy_b128(d0, s0);
        async_copy_b128(d0 + 8, s0 + 8);
      } else {
        v8h z = {};
        *(v8h*)d0 = z;
        *(v8h*)(d0 + 8) = z;
      }
    }
    // stage B tile (async copy from transposed weights)
    {
      _Float16* d0 = &lsb[srow * 32 + sseg];
      const _Float16* s0 = WT + (size_t)(bn + srow) * K + k0 + sseg;
      async_copy_b128(d0, s0);
      async_copy_b128(d0 + 8, s0 + 8);
    }
    wait_async();
    __syncthreads();

    // A fragment: halves 0..7 -> K = ka..ka+7, halves 8..15 -> K = ka+16..+23
    // with ka = (lane/16)*8  (ISA 16-bit A 16x32 layout)
    const int ka = (lane >> 4) * 8;
    // B fragment: halves 0..7 -> K = kb..kb+7, halves 8..15 -> K = kb+16..+23, kb = (lane/16)*8
    const int kb = (lane >> 4) * 8;
    v16h af[2], bf[4];
#pragma unroll
    for (int i = 0; i < 2; ++i) {
      int row = wm * 32 + i * 16 + (lane & 15);
      v8h lo = *(const v8h*)&lsa[row * 32 + ka];
      v8h hi = *(const v8h*)&lsa[row * 32 + ka + 16];
      af[i] = __builtin_shufflevector(lo, hi, 0, 1, 2, 3, 4, 5, 6, 7, 8, 9, 10,
                                      11, 12, 13, 14, 15);
    }
#pragma unroll
    for (int j = 0; j < 4; ++j) {
      int col = wn * 64 + j * 16 + (lane & 15);
      v8h lo = *(const v8h*)&lsb[col * 32 + kb];
      v8h hi = *(const v8h*)&lsb[col * 32 + kb + 16];
      bf[j] = __builtin_shufflevector(lo, hi, 0, 1, 2, 3, 4, 5, 6, 7, 8, 9, 10,
                                      11, 12, 13, 14, 15);
    }
#pragma unroll
    for (int i = 0; i < 2; ++i)
#pragma unroll
      for (int j = 0; j < 4; ++j)
        acc[i][j] = __builtin_amdgcn_wmma_f32_16x16x32_f16(
            false, af[i], false, bf[j], (short)0, acc[i][j], false, false);
    __syncthreads();
  }

  // epilogue: C/D layout -> M = v + 8*(lane/16), N = lane%16
#pragma unroll
  for (int i = 0; i < 2; ++i)
#pragma unroll
    for (int j = 0; j < 4; ++j)
#pragma unroll
      for (int v = 0; v < 8; ++v) {
        int row = bm + wm * 32 + i * 16 + v + ((lane >> 4) << 3);
        int col = bn + wn * 64 + j * 16 + (lane & 15);
        if (row < M) {
          float x = acc[i][j][v];
          if (bias) x += bias[col];
          if (GELU) {
            float x3 = x * x * x;
            x = 0.5f * x *
                (1.0f + tanhf(0.7978845608028654f * (x + 0.044715f * x3)));
          }
          if (RESID) x += resid[(size_t)row * N + col];
          if (OUT_F16)
            ((_Float16*)Cout)[(size_t)row * N + col] = (_Float16)x;
          else
            ((float*)Cout)[(size_t)row * N + col] = x;
        }
      }
}

// ------------------------------ LN kernels ---------------------------------
__device__ __forceinline__ void block_ln_stats(float a0, float a1, float& mean,
                                               float& rs) {
  float s = a0 + a1, s2 = a0 * a0 + a1 * a1;
  for (int m = 16; m; m >>= 1) {
    s += __shfl_xor(s, m, 32);
    s2 += __shfl_xor(s2, m, 32);
  }
  __shared__ float ws1[8], ws2[8];
  int tid = threadIdx.x;
  if ((tid & 31) == 0) {
    ws1[tid >> 5] = s;
    ws2[tid >> 5] = s2;
  }
  __syncthreads();
  float ts = 0.f, ts2 = 0.f;
#pragma unroll
  for (int i = 0; i < 8; ++i) {
    ts += ws1[i];
    ts2 += ws2[i];
  }
  mean = ts * (1.0f / 512.0f);
  float var = ts2 * (1.0f / 512.0f) - mean * mean;
  rs = rsqrtf(var + 1e-5f);
  __syncthreads();
}

// LN over D=512, f32 in -> f16 out. one block per row.
__global__ __launch_bounds__(256) void ln_f16_kernel(
    const float* __restrict__ x, const float* __restrict__ g,
    const float* __restrict__ bb, _Float16* __restrict__ out, int M) {
  int row = blockIdx.x;
  if (row >= M) return;
  int tid = threadIdx.x;
  const float* xr = x + (size_t)row * 512;
  float a0 = xr[tid], a1 = xr[tid + 256];
  float mean, rs;
  block_ln_stats(a0, a1, mean, rs);
  out[(size_t)row * 512 + tid] = (_Float16)((a0 - mean) * rs * g[tid] + bb[tid]);
  out[(size_t)row * 512 + tid + 256] =
      (_Float16)((a1 - mean) * rs * g[tid + 256] + bb[tid + 256]);
}

// gather (unit_emb[su]+type_emb[sty]) then LN(enc_lnk) -> f16 [40960,512]
__global__ __launch_bounds__(256) void gather_ln_inputs_kernel(
    const int* __restrict__ su, const int* __restrict__ sty,
    const float* __restrict__ unit_emb, const float* __restrict__ type_emb,
    const float* __restrict__ g, const float* __restrict__ bb,
    _Float16* __restrict__ out) {
  int row = blockIdx.x;  // 0..40959
  int tid = threadIdx.x;
  int u = su[row], ty = sty[row];
  const float* ue = unit_emb + (size_t)u * 512;
  const float* te = type_emb + (size_t)ty * 512;
  float a0 = ue[tid] + te[tid];
  float a1 = ue[tid + 256] + te[tid + 256];
  float mean, rs;
  block_ln_stats(a0, a1, mean, rs);
  out[(size_t)row * 512 + tid] = (_Float16)((a0 - mean) * rs * g[tid] + bb[tid]);
  out[(size_t)row * 512 + tid + 256] =
      (_Float16)((a1 - mean) * rs * g[tid + 256] + bb[tid + 256]);
}

// ---------------------------- small init kernels ---------------------------
__global__ void lat_init_kernel(const int* __restrict__ latent_index,
                                const float* __restrict__ latent_emb,
                                float* __restrict__ lat) {
  int idx = blockIdx.x * 256 + threadIdx.x;  // 1280*512
  if (idx >= PBT * PNL * PDIM) return;
  int row = idx >> 9, d = idx & 511;
  int r = row >> 3, l = row & 7;
  int li = latent_index[(r & 7) * PNL + l];  // tile(latents): row -> latents[r%B]
  lat[idx] = latent_emb[(size_t)li * 512 + d];
}

__global__ void outq_init_kernel(const int* __restrict__ odi,
                                 const int* __restrict__ osi,
                                 const float* __restrict__ task_emb,
                                 const float* __restrict__ sess_emb,
                                 float* __restrict__ outq) {
  int idx = blockIdx.x * 256 + threadIdx.x;  // 1024*512
  if (idx >= PB * PNOUT * PDIM) return;
  int ro = idx >> 9, d = idx & 511;
  outq[idx] = task_emb[(size_t)odi[ro] * 512 + d] +
              sess_emb[(size_t)osi[ro] * 512 + d];
}

__global__ void f32_to_f16_kernel(const float* __restrict__ in,
                                  _Float16* __restrict__ out, int n) {
  for (int i = blockIdx.x * blockDim.x + threadIdx.x; i < n;
       i += gridDim.x * blockDim.x)
    out[i] = (_Float16)in[i];
}

// W [K,N] f32 -> WT [N,K] f16  (transpose + convert)
__global__ void wT_f16_kernel(const float* __restrict__ in,
                              _Float16* __restrict__ out, int K, int N) {
  int total = K * N;
  for (int idx = blockIdx.x * 256 + threadIdx.x; idx < total;
       idx += gridDim.x * 256) {
    int n = idx / K, k = idx - n * K;
    out[idx] = (_Float16)in[(size_t)k * N + n];
  }
}

// --------------------------- encoder cross-attn ----------------------------
// block = one row r (of BT). 8 latent queries x 256 keys x 8 heads.
// Q f32 [1280,512]; KV f16 [40960,1024] (k|v). input_mask is all-true.
__global__ __launch_bounds__(256) void enc_attn_kernel(
    const float* __restrict__ Q, const _Float16* __restrict__ KV,
    const float* __restrict__ lat_ts, const float* __restrict__ spk_ts,
    _Float16* __restrict__ O) {
  int r = blockIdx.x;
  int tid = threadIdx.x, lane = tid & 31, wave = tid >> 5;
  __shared__ float qs[8][64];
  __shared__ float sc[8][256];
  __shared__ _Float16 vs[256][64];

  for (int h = 0; h < 8; ++h) {
    {  // rope queries: wave = query l, lane = pair p
      int l = wave, p = lane;
      float c, s;
      rope_cs(lat_ts[l], p, c, s);
      const float* qp = Q + (size_t)(r * 8 + l) * 512 + h * 64 + 2 * p;
      float x0 = qp[0], x1 = qp[1];
      qs[l][2 * p] = x0 * c - x1 * s;
      qs[l][2 * p + 1] = x1 * c + x0 * s;
    }
    __syncthreads();
    {  // each thread = one key j: rope k/v, scores vs all 8 queries
      int j = tid;
      float t = spk_ts[(size_t)r * 256 + j];
      const _Float16* kv = KV + (size_t)(r * 256 + j) * 1024;
      float acc[8];
#pragma unroll
      for (int l = 0; l < 8; ++l) acc[l] = 0.f;
      for (int p = 0; p < 32; ++p) {
        float c, s;
        rope_cs(t, p, c, s);
        float k0 = (float)kv[h * 64 + 2 * p], k1 = (float)kv[h * 64 + 2 * p + 1];
        float kr0 = k0 * c - k1 * s, kr1 = k1 * c + k0 * s;
        float v0 = (float)kv[512 + h * 64 + 2 * p];
        float v1 = (float)kv[512 + h * 64 + 2 * p + 1];
        vs[j][2 * p] = (_Float16)(v0 * c - v1 * s);
        vs[j][2 * p + 1] = (_Float16)(v1 * c + v0 * s);
#pragma unroll
        for (int l = 0; l < 8; ++l)
          acc[l] += kr0 * qs[l][2 * p] + kr1 * qs[l][2 * p + 1];
      }
#pragma unroll
      for (int l = 0; l < 8; ++l) sc[l][j] = acc[l] * 0.125f;  // 1/sqrt(64)
    }
    __syncthreads();
    {  // softmax: wave l over 256 keys
      int l = wave;
      float vals[8], mx = -1e30f;
#pragma unroll
      for (int i = 0; i < 8; ++i) {
        vals[i] = sc[l][lane + 32 * i];
        mx = fmaxf(mx, vals[i]);
      }
      for (int m = 16; m; m >>= 1) mx = fmaxf(mx, __shfl_xor(mx, m, 32));
      float sum = 0.f;
#pragma unroll
      for (int i = 0; i < 8; ++i) {
        vals[i] = __expf(vals[i] - mx);
        sum += vals[i];
      }
      for (int m = 16; m; m >>= 1) sum += __shfl_xor(sum, m, 32);
      float inv = 1.f / sum;
#pragma unroll
      for (int i = 0; i < 8; ++i) sc[l][lane + 32 * i] = vals[i] * inv;
    }
    __syncthreads();
    {  // output + un-rotate with query angle
      int l = wave, p = lane;
      float o0 = 0.f, o1 = 0.f;
      for (int j = 0; j < 256; ++j) {
        float a = sc[l][j];
        o0 += a * (float)vs[j][2 * p];
        o1 += a * (float)vs[j][2 * p + 1];
      }
      float c, s;
      rope_cs(lat_ts[l], p, c, s);
      _Float16* op = O + (size_t)(r * 8 + l) * 512 + h * 64 + 2 * p;
      op[0] = (_Float16)(o0 * c + o1 * s);
      op[1] = (_Float16)(o1 * c - o0 * s);
    }
    __syncthreads();
  }
}

// ----------------------------- self attention ------------------------------
// block = (r, head). 8 tokens, full attention.
__global__ __launch_bounds__(256) void self_attn_kernel(
    const float* __restrict__ QKV, const float* __restrict__ lat_ts,
    _Float16* __restrict__ O) {
  int r = blockIdx.x, h = blockIdx.y;
  int tid = threadIdx.x;
  __shared__ float qs[8][64], ks[8][64], vsm[8][64], sc[8][8], aw[8][8];
  int l = tid >> 5, p = tid & 31;
  float c, s;
  rope_cs(lat_ts[l], p, c, s);
  const float* base = QKV + (size_t)(r * 8 + l) * 1536;
  float q0 = base[h * 64 + 2 * p], q1 = base[h * 64 + 2 * p + 1];
  float k0 = base[512 + h * 64 + 2 * p], k1 = base[512 + h * 64 + 2 * p + 1];
  float v0 = base[1024 + h * 64 + 2 * p], v1 = base[1024 + h * 64 + 2 * p + 1];
  qs[l][2 * p] = q0 * c - q1 * s;
  qs[l][2 * p + 1] = q1 * c + q0 * s;
  ks[l][2 * p] = k0 * c - k1 * s;
  ks[l][2 * p + 1] = k1 * c + k0 * s;
  vsm[l][2 * p] = v0 * c - v1 * s;
  vsm[l][2 * p + 1] = v1 * c + v0 * s;
  __syncthreads();
  if (tid < 64) {
    int lq = tid >> 3, lk = tid & 7;
    float a = 0.f;
#pragma unroll
    for (int d = 0; d < 64; ++d) a += qs[lq][d] * ks[lk][d];
    sc[lq][lk] = a * 0.125f;
  }
  __syncthreads();
  if (tid < 8) {
    float mx = -1e30f;
#pragma unroll
    for (int k = 0; k < 8; ++k) mx = fmaxf(mx, sc[tid][k]);
    float sum = 0.f;
#pragma unroll
    for (int k = 0; k < 8; ++k) {
      float e = __expf(sc[tid][k] - mx);
      aw[tid][k] = e;
      sum += e;
    }
    float inv = 1.f / sum;
#pragma unroll
    for (int k = 0; k < 8; ++k) aw[tid][k] *= inv;
  }
  __syncthreads();
  {
    float o0 = 0.f, o1 = 0.f;
#pragma unroll
    for (int k = 0; k < 8; ++k) {
      float a = aw[l][k];
      o0 += a * vsm[k][2 * p];
      o1 += a * vsm[k][2 * p + 1];
    }
    _Float16* op = O + (size_t)(r * 8 + l) * 512 + h * 64 + 2 * p;
    op[0] = (_Float16)(o0 * c + o1 * s);
    op[1] = (_Float16)(o1 * c - o0 * s);
  }
}

// ---------------------------- decoder attention ----------------------------
// block = one output query (b*128+o); wave = head; sliding window of <=3 bins.
__global__ __launch_bounds__(256) void dec_attn_kernel(
    const float* __restrict__ Qd, const float* __restrict__ KVd,
    const float* __restrict__ out_ts, const int* __restrict__ out_bin,
    _Float16* __restrict__ O) {
  int ro = blockIdx.x;
  int b = ro >> 7;
  int h = threadIdx.x >> 5, lane = threadIdx.x & 31;
  float tq = out_ts[ro];
  int bidx = out_bin[ro];
  if (bidx < 1) bidx = 1;
  int t0 = bidx - 3;
  if (t0 < 0) t0 = 0;
  int t1 = bidx < PT ? bidx : PT;
  int nt = t1 - t0;

  float cq, sq;
  rope_cs(tq, lane, cq, sq);
  const float* qp = Qd + (size_t)ro * 512 + h * 64 + 2 * lane;
  float q0r = qp[0] * cq - qp[1] * sq;
  float q1r = qp[1] * cq + qp[0] * sq;

  float scv[3], vr0[3], vr1[3];
#pragma unroll
  for (int i = 0; i < 3; ++i) {
    bool act = i < nt;
    int t = t0 + (act ? i : 0);
    float tk = (float)(t + 1) * 0.05f;
    float ck, sk;
    rope_cs(tk, lane, ck, sk);
    const float* kv = KVd + (size_t)(b * PT + t) * 1024;
    float k0 = kv[h * 64 + 2 * lane], k1 = kv[h * 64 + 2 * lane + 1];
    float kr0 = k0 * ck - k1 * sk, kr1 = k1 * ck + k0 * sk;
    float v0 = kv[512 + h * 64 + 2 * lane], v1 = kv[512 + h * 64 + 2 * lane + 1];
    vr0[i] = act ? (v0 * ck - v1 * sk) : 0.f;
    vr1[i] = act ? (v1 * ck + v0 * sk) : 0.f;
    float part = q0r * kr0 + q1r * kr1;
    for (int m = 16; m; m >>= 1) part += __shfl_xor(part, m, 32);
    scv[i] = act ? part * 0.125f : -1e30f;
  }
  float mx = fmaxf(scv[0], fmaxf(scv[1], scv[2]));
  float sum = 0.f;
#pragma unroll
  for (int i = 0; i < 3; ++i) {
    scv[i] = __expf(scv[i] - mx);
    sum += scv[i];
  }
  float inv = 1.f / sum;
  float o0 = 0.f, o1 = 0.f;
#pragma unroll
  for (int i = 0; i < 3; ++i) {
    float a = scv[i] * inv;
    o0 += a * vr0[i];
    o1 += a * vr1[i];
  }
  _Float16* op = O + (size_t)ro * 512 + h * 64 + 2 * lane;
  op[0] = (_Float16)(o0 * cq + o1 * sq);
  op[1] = (_Float16)(o1 * cq - o0 * sq);
}

// --------------------------------- GRU step --------------------------------
// one launch per time step; thread = (b, j) computes recurrent gates.
__global__ __launch_bounds__(256) void gru_step_kernel(
    const float* __restrict__ XP, const float* __restrict__ whh,
    const float* __restrict__ bhh, float* __restrict__ H, int t) {
  int idx = blockIdx.x * 256 + threadIdx.x;  // 0..4095
  if (idx >= PB * PRNN) return;
  int b = idx >> 9, j = idx & 511;
  const float* hprev = (t == 0) ? nullptr : H + (size_t)(b * PT + t - 1) * 512;
  float hr = bhh[j], hz = bhh[512 + j], hn = bhh[1024 + j];
  if (hprev) {
    const float* wr = whh + (size_t)j * 512;
    const float* wz = whh + (size_t)(512 + j) * 512;
    const float* wn = whh + (size_t)(1024 + j) * 512;
    for (int d = 0; d < 512; ++d) {
      float hv = hprev[d];
      hr += wr[d] * hv;
      hz += wz[d] * hv;
      hn += wn[d] * hv;
    }
  }
  const float* xp = XP + (size_t)(b * PT + t) * 1536;
  float rg = 1.f / (1.f + __expf(-(xp[j] + hr)));
  float zg = 1.f / (1.f + __expf(-(xp[512 + j] + hz)));
  float ng = tanhf(xp[1024 + j] + rg * hn);
  float hp = (t == 0) ? 0.f : hprev[j];
  H[(size_t)(b * PT + t) * 512 + j] = (1.f - zg) * ng + zg * hp;
}

// --------------------------------- readout ---------------------------------
__global__ void readout_kernel(const float* __restrict__ ol,
                               const float* __restrict__ w,
                               const float* __restrict__ bias,
                               float* __restrict__ out) {
  int idx = blockIdx.x * 256 + threadIdx.x;  // 1024*2
  if (idx >= PB * PNOUT * 2) return;
  int ro = idx >> 1, c = idx & 1;
  float a = bias[c];
  const float* x = ol + (size_t)ro * 512;
  for (int d = 0; d < 512; ++d) a += x[d] * w[d * 2 + c];
  out[idx] = a;
}

// ------------------------------- host driver -------------------------------
static inline void run_gemm(const _Float16* A, const _Float16* WT,
                            const float* bias, const float* resid, void* C,
                            int M, int N, int K, int outF16, int gelu,
                            hipStream_t s) {
  dim3 g((M + 127) / 128, N / 128);
  if (outF16 && gelu)
    gemm_wmma<1, 1, 0><<<g, 256, 0, s>>>(A, WT, bias, resid, C, M, N, K);
  else if (outF16)
    gemm_wmma<1, 0, 0><<<g, 256, 0, s>>>(A, WT, bias, resid, C, M, N, K);
  else if (resid)
    gemm_wmma<0, 0, 1><<<g, 256, 0, s>>>(A, WT, bias, resid, C, M, N, K);
  else
    gemm_wmma<0, 0, 0><<<g, 256, 0, s>>>(A, WT, bias, resid, C, M, N, K);
}

extern "C" void kernel_launch(void* const* d_in, const int* in_sizes, int n_in,
                              void* d_out, int out_size, void* d_ws,
                              size_t ws_size, hipStream_t stream) {
  (void)in_sizes; (void)n_in; (void)out_size; (void)ws_size;
  // ---- data inputs (setup_inputs order) ----
  const int*   spike_unit = (const int*)d_in[0];
  const float* spike_ts   = (const float*)d_in[1];
  const int*   spike_type = (const int*)d_in[2];
  // d_in[3] input_mask (all true) and d_in[4] n_intervals are unused.
  const int*   latent_index = (const int*)d_in[5];
  const float* latent_ts    = (const float*)d_in[6];  // row 0 used (bcast)
  const int*   out_sess = (const int*)d_in[7];
  const float* out_ts   = (const float*)d_in[8];
  const int*   out_dec  = (const int*)d_in[9];
  const int*   out_bin  = (const int*)d_in[10];
  // ---- params: pytree leaves, sorted by key, starting at index 11 ----
  const int P = 11;
  auto F = [&](int k) { return (const float*)d_in[P + k]; };
  const float *dec_bo = F(0), *dec_ff_b1 = F(1), *dec_ff_b2 = F(2),
              *dec_ff_ln_b = F(3), *dec_ff_ln_g = F(4), *dec_ff_w1 = F(5),
              *dec_ff_w2 = F(6), *dec_lnk_b = F(7), *dec_lnk_g = F(8),
              *dec_lnq_b = F(9), *dec_lnq_g = F(10), *dec_wkv = F(11),
              *dec_wo = F(12), *dec_wq = F(13);
  const float *enc_bo = F(14), *enc_ff_b1 = F(15), *enc_ff_b2 = F(16),
              *enc_ff_ln_b = F(17), *enc_ff_ln_g = F(18), *enc_ff_w1 = F(19),
              *enc_ff_w2 = F(20), *enc_lnk_b = F(21), *enc_lnk_g = F(22),
              *enc_lnq_b = F(23), *enc_lnq_g = F(24), *enc_wkv = F(25),
              *enc_wo = F(26), *enc_wq = F(27);
  const float *gru_bhh = F(28), *gru_bih = F(29), *gru_whh = F(30),
              *gru_wih = F(31), *latent_emb = F(32);
  const float* p_ff_b1[2]   = {F(33), F(44)};
  const float* p_ff_b2[2]   = {F(34), F(45)};
  const float* p_ff_ln_b[2] = {F(35), F(46)};
  const float* p_ff_ln_g[2] = {F(36), F(47)};
  const float* p_ff_w1[2]   = {F(37), F(48)};
  const float* p_ff_w2[2]   = {F(38), F(49)};
  const float* p_sa_bo[2]   = {F(39), F(50)};
  const float* p_sa_ln_b[2] = {F(40), F(51)};
  const float* p_sa_ln_g[2] = {F(41), F(52)};
  const float* p_sa_wo[2]   = {F(42), F(53)};
  const float* p_sa_wqkv[2] = {F(43), F(54)};
  const float *readout_b = F(55), *readout_w = F(56), *session_emb = F(57),
              *task_emb = F(58), *type_emb = F(59), *unit_emb = F(60);

  // ---- workspace arena ----
  char* ws = (char*)d_ws;
  size_t off = 0;
  auto alloc = [&](size_t bytes) {
    size_t r = off;
    off += (bytes + 255) & ~(size_t)255;
    return r;
  };
  auto A16 = [&](size_t n) { return (_Float16*)(ws + alloc(n * 2)); };
  auto A32 = [&](size_t n) { return (float*)(ws + alloc(n * 4)); };

  // f16 transposed weights [N,K]
  _Float16* w_enc_wkv = A16(512 * 1024);
  _Float16* w_enc_wq  = A16(512 * 512);
  _Float16* w_enc_wo  = A16(512 * 512);
  _Float16* w_enc_w1  = A16(512 * 2048);
  _Float16* w_enc_w2  = A16(2048 * 512);
  _Float16 *w_sa_wqkv[2], *w_sa_wo[2], *w_ff_w1[2], *w_ff_w2[2];
  for (int i = 0; i < 2; ++i) {
    w_sa_wqkv[i] = A16(512 * 1536);
    w_sa_wo[i]   = A16(512 * 512);
    w_ff_w1[i]   = A16(512 * 2048);
    w_ff_w2[i]   = A16(2048 * 512);
  }
  _Float16* w_dec_wq  = A16(512 * 512);
  _Float16* w_dec_wkv = A16(512 * 1024);
  _Float16* w_dec_wo  = A16(512 * 512);
  _Float16* w_dec_w1  = A16(512 * 2048);
  _Float16* w_dec_w2  = A16(2048 * 512);
  _Float16* w_gruT    = A16(4096 * 1536);  // gru_wih is already [N=1536,K=4096]
  // activations
  float*    lat   = A32(PBT * PNL * PDIM);          // [1280,512]
  _Float16* Xkv16 = A16((size_t)PBT * PNIN * 512);  // [40960,512]
  _Float16* KV16  = A16((size_t)PBT * PNIN * 1024); // [40960,1024]
  _Float16* ln16  = A16(PBT * PNL * PDIM);          // [1280,512] (reused)
  float*    Qbuf  = A32(PBT * PNL * PDIM);          // enc Q / dec Qd
  float*    QKV   = A32(PBT * PNL * 1536);
  _Float16* O16   = A16(PBT * PNL * PDIM);
  _Float16* FFH16 = A16(PBT * PNL * 2048);
  float*    XP    = A32(PBT * 1536);
  float*    H     = A32(PBT * PRNN);
  float*    outq  = A32(PB * PNOUT * PDIM);
  float*    ldec  = A32(PB * PNOUT * PDIM);
  float*    oldec = A32(PB * PNOUT * PDIM);
  float*    KVd   = A32(PBT * 1024);

  // ---- weight conversion: W [K,N] f32 -> WT [N,K] f16 ----
  auto cvtT = [&](const float* src, _Float16* dst, int K, int N) {
    wT_f16_kernel<<<1024, 256, 0, stream>>>(src, dst, K, N);
  };
  cvtT(enc_wkv, w_enc_wkv, 512, 1024);
  cvtT(enc_wq, w_enc_wq, 512, 512);
  cvtT(enc_wo, w_enc_wo, 512, 512);
  cvtT(enc_ff_w1, w_enc_w1, 512, 2048);
  cvtT(enc_ff_w2, w_enc_w2, 2048, 512);
  for (int i = 0; i < 2; ++i) {
    cvtT(p_sa_wqkv[i], w_sa_wqkv[i], 512, 1536);
    cvtT(p_sa_wo[i], w_sa_wo[i], 512, 512);
    cvtT(p_ff_w1[i], w_ff_w1[i], 512, 2048);
    cvtT(p_ff_w2[i], w_ff_w2[i], 2048, 512);
  }
  cvtT(dec_wq, w_dec_wq, 512, 512);
  cvtT(dec_wkv, w_dec_wkv, 512, 1024);
  cvtT(dec_wo, w_dec_wo, 512, 512);
  cvtT(dec_ff_w1, w_dec_w1, 512, 2048);
  cvtT(dec_ff_w2, w_dec_w2, 2048, 512);
  f32_to_f16_kernel<<<1024, 256, 0, stream>>>(gru_wih, w_gruT, 1536 * 4096);

  // ---- encoder: inputs -> KV ----
  gather_ln_inputs_kernel<<<PBT * PNIN, 256, 0, stream>>>(
      spike_unit, spike_type, unit_emb, type_emb, enc_lnk_g, enc_lnk_b, Xkv16);
  run_gemm(Xkv16, w_enc_wkv, nullptr, nullptr, KV16, PBT * PNIN, 1024, 512,
           /*f16*/ 1, 0, stream);

  // lat init + cross-attention
  lat_init_kernel<<<(PBT * PNL * PDIM + 255) / 256, 256, 0, stream>>>(
      latent_index, latent_emb, lat);
  ln_f16_kernel<<<PBT * PNL, 256, 0, stream>>>(lat, enc_lnq_g, enc_lnq_b, ln16,
                                               PBT * PNL);
  run_gemm(ln16, w_enc_wq, nullptr, nullptr, Qbuf, PBT * PNL, 512, 512, 0, 0,
           stream);
  enc_attn_kernel<<<PBT, 256, 0, stream>>>(Qbuf, KV16, latent_ts, spike_ts,
                                           O16);
  run_gemm(O16, w_enc_wo, enc_bo, lat, lat, PBT * PNL, 512, 512, 0, 0, stream);

  // ---- process layers: self-attn + FFN (x2), then enc_ff ----
  for (int i = 0; i < 2; ++i) {
    ln_f16_kernel<<<PBT * PNL, 256, 0, stream>>>(lat, p_sa_ln_g[i],
                                                 p_sa_ln_b[i], ln16, PBT * PNL);
    run_gemm(ln16, w_sa_wqkv[i], nullptr, nullptr, QKV, PBT * PNL, 1536, 512, 0,
             0, stream);
    self_attn_kernel<<<dim3(PBT, 8), 256, 0, stream>>>(QKV, latent_ts, O16);
    run_gemm(O16, w_sa_wo[i], p_sa_bo[i], lat, lat, PBT * PNL, 512, 512, 0, 0,
             stream);
    ln_f16_kernel<<<PBT * PNL, 256, 0, stream>>>(lat, p_ff_ln_g[i],
                                                 p_ff_ln_b[i], ln16, PBT * PNL);
    run_gemm(ln16, w_ff_w1[i], p_ff_b1[i], nullptr, FFH16, PBT * PNL, 2048, 512,
             1, /*gelu*/ 1, stream);
    run_gemm(FFH16, w_ff_w2[i], p_ff_b2[i], lat, lat, PBT * PNL, 512, 2048, 0,
             0, stream);
  }
  ln_f16_kernel<<<PBT * PNL, 256, 0, stream>>>(lat, enc_ff_ln_g, enc_ff_ln_b,
                                               ln16, PBT * PNL);
  run_gemm(ln16, w_enc_w1, enc_ff_b1, nullptr, FFH16, PBT * PNL, 2048, 512, 1,
           1, stream);
  run_gemm(FFH16, w_enc_w2, enc_ff_b2, lat, lat, PBT * PNL, 512, 2048, 0, 0,
           stream);

  // ---- GRU: seq = lat viewed as [160,4096] ----
  f32_to_f16_kernel<<<1024, 256, 0, stream>>>(lat, ln16, PBT * PNL * PDIM);
  run_gemm(ln16, w_gruT, gru_bih, nullptr, XP, PBT, 1536, 4096, 0, 0, stream);
  for (int t = 0; t < PT; ++t)
    gru_step_kernel<<<(PB * PRNN + 255) / 256, 256, 0, stream>>>(XP, gru_whh,
                                                                 gru_bhh, H, t);

  // ---- decoder ----
  outq_init_kernel<<<(PB * PNOUT * PDIM + 255) / 256, 256, 0, stream>>>(
      out_dec, out_sess, task_emb, session_emb, outq);
  ln_f16_kernel<<<PB * PNOUT, 256, 0, stream>>>(outq, dec_lnq_g, dec_lnq_b,
                                                ln16, PB * PNOUT);
  run_gemm(ln16, w_dec_wq, nullptr, nullptr, Qbuf, PB * PNOUT, 512, 512, 0, 0,
           stream);
  ln_f16_kernel<<<PBT, 256, 0, stream>>>(H, dec_lnk_g, dec_lnk_b, ln16, PBT);
  run_gemm(ln16, w_dec_wkv, nullptr, nullptr, KVd, PBT, 1024, 512, 0, 0,
           stream);
  dec_attn_kernel<<<PB * PNOUT, 256, 0, stream>>>(Qbuf, KVd, out_ts, out_bin,
                                                  O16);
  run_gemm(O16, w_dec_wo, dec_bo, outq, ldec, PB * PNOUT, 512, 512, 0, 0,
           stream);
  ln_f16_kernel<<<PB * PNOUT, 256, 0, stream>>>(ldec, dec_ff_ln_g, dec_ff_ln_b,
                                                ln16, PB * PNOUT);
  run_gemm(ln16, w_dec_w1, dec_ff_b1, nullptr, FFH16, PB * PNOUT, 2048, 512, 1,
           1, stream);
  run_gemm(FFH16, w_dec_w2, dec_ff_b2, ldec, oldec, PB * PNOUT, 512, 2048, 0, 0,
           stream);
  readout_kernel<<<(PB * PNOUT * 2 + 255) / 256, 256, 0, stream>>>(
      oldec, readout_w, readout_b, (float*)d_out);
}
